// CobafaFeatureField_15668040695867
// MI455X (gfx1250) — matrix operand order repaired and checked by reading, not compiled, not run on hardware
//
#include <hip/hip_runtime.h>

typedef _Float16 v16h __attribute__((ext_vector_type(16)));
typedef _Float16 v8h  __attribute__((ext_vector_type(8)));
typedef float    v8f  __attribute__((ext_vector_type(8)));

union V16U { v16h v; v8h h[2]; };

__device__ __forceinline__ v16h mk16(v8h lo, v8h hi) {
  V16U u; u.h[0] = lo; u.h[1] = hi; return u.v;
}

// 16-byte-granule XOR swizzle over a [rows x 128] f16 buffer: the 16 lanes of a
// WMMA fragment read (one 16B chunk per row, 16 distinct rows) hit 16 distinct
// LDS bank groups.
__device__ __forceinline__ int swz8(int row, int chunk) {   // chunk-granular (v8h)
  return row * 128 + ((chunk ^ (row & 15)) << 3);
}
__device__ __forceinline__ int swzc(int row, int col) {     // element-granular
  return row * 128 + ((((col >> 3) ^ (row & 15)) << 3) | (col & 7));
}

struct Params {
  const float* x;
  const float* g0; const float* g1; const float* g2; const float* g3;
  const float* coef;
  const float* W[7];
  const float* Bv[7];
  float* out;
};

// align_corners trilinear gather of a [4, R, R, R] grid at (px,py,pz) in [-1,1]
template<int R>
__device__ __forceinline__ void trilerp4(const float* __restrict__ g,
                                         float px, float py, float pz,
                                         float* __restrict__ out4) {
  const float s = 0.5f * (float)(R - 1);
  float cx = (px + 1.0f) * s, cy = (py + 1.0f) * s, cz = (pz + 1.0f) * s;
  float fxf = floorf(cx), fyf = floorf(cy), fzf = floorf(cz);
  float fx = cx - fxf, fy = cy - fyf, fz = cz - fzf;
  int x0 = (int)fxf, y0 = (int)fyf, z0 = (int)fzf;
  x0 = min(max(x0, 0), R - 1);
  y0 = min(max(y0, 0), R - 1);
  z0 = min(max(z0, 0), R - 1);
  int x1 = min(x0 + 1, R - 1), y1 = min(y0 + 1, R - 1), z1 = min(z0 + 1, R - 1);

  const int R2 = R * R;
  int b00 = z0 * R2 + y0 * R;
  int b01 = z0 * R2 + y1 * R;
  int b10 = z1 * R2 + y0 * R;
  int b11 = z1 * R2 + y1 * R;
  int i000 = b00 + x0, i001 = b00 + x1, i010 = b01 + x0, i011 = b01 + x1;
  int i100 = b10 + x0, i101 = b10 + x1, i110 = b11 + x0, i111 = b11 + x1;

  float gx0 = 1.0f - fx, gy0 = 1.0f - fy, gz0 = 1.0f - fz;
  float w000 = gz0 * gy0 * gx0, w001 = gz0 * gy0 * fx;
  float w010 = gz0 * fy  * gx0, w011 = gz0 * fy  * fx;
  float w100 = fz  * gy0 * gx0, w101 = fz  * gy0 * fx;
  float w110 = fz  * fy  * gx0, w111 = fz  * fy  * fx;

#pragma unroll
  for (int c = 0; c < 4; ++c) {
    const float* gc = g + (size_t)c * R * R * R;
    out4[c] = gc[i000] * w000 + gc[i001] * w001 + gc[i010] * w010 + gc[i011] * w011
            + gc[i100] * w100 + gc[i101] * w101 + gc[i110] * w110 + gc[i111] * w111;
  }
}

__device__ __forceinline__ float enc1(float v, float f) {
  float t = v * f;
  return 2.0f * (t - floorf(t)) - 1.0f;   // 2*(f*x mod 1) - 1, mod in [0,1)
}

// Both of this wave's 16-point tiles through one MLP layer, sharing B fragments.
// Activations LDS: [256 x 128] f16 rows (WMMA A layout per lane).
// Weights LDS: Wt[n][k] f16 (B layout: same k order as A, two 8-k chunks per lane).
template<int NKC, bool RELU, bool TO_GLOBAL>
__device__ __forceinline__ void mlp_layer_pair(_Float16* __restrict__ s_act,
                                               const _Float16* __restrict__ s_w,
                                               const float* __restrict__ Bj,
                                               float* __restrict__ outp,
                                               int t0, int ln, int sel) {
  // Load A fragments for both tiles (all K chunks) before any in-place store.
  v16h a0[NKC], a1[NKC];
  const int row0 = t0 * 16 + ln;
  const int row1 = row0 + 16;
#pragma unroll
  for (int kc = 0; kc < NKC; ++kc) {
    // lane<16 needs K cols kc*32+[0..7] & [16..23]; lane>=16: [8..15] & [24..31]
    a0[kc] = mk16(*(const v8h*)(s_act + swz8(row0, kc * 4 + sel)),
                  *(const v8h*)(s_act + swz8(row0, kc * 4 + sel + 2)));
    a1[kc] = mk16(*(const v8h*)(s_act + swz8(row1, kc * 4 + sel)),
                  *(const v8h*)(s_act + swz8(row1, kc * 4 + sel + 2)));
  }

#pragma unroll
  for (int nt = 0; nt < 8; ++nt) {
    // One shared set of B fragments for this N-tile (used by both point tiles).
    v16h bfr[NKC];
    const int wrow = nt * 16 + ln;
#pragma unroll
    for (int kc = 0; kc < NKC; ++kc) {
      // B: lane<16 -> K cols kc*32+[0..7] & [16..23]; lane>=16 -> [8..15] & [24..31]
      bfr[kc] = mk16(*(const v8h*)(s_w + swz8(wrow, kc * 4 + sel)),
                     *(const v8h*)(s_w + swz8(wrow, kc * 4 + sel + 2)));
    }

    // Two independent accumulation chains -> ILP to hide ds latency + hazards.
    v8f acc0 = {}, acc1 = {};
#pragma unroll
    for (int kc = 0; kc < NKC; ++kc) {
      acc0 = __builtin_amdgcn_wmma_f32_16x16x32_f16(
          false, a0[kc], false, bfr[kc], (short)0, acc0, false, false);
      acc1 = __builtin_amdgcn_wmma_f32_16x16x32_f16(
          false, a1[kc], false, bfr[kc], (short)0, acc1, false, false);
    }

    const float bias = Bj[nt * 16 + ln];
#pragma unroll
    for (int r = 0; r < 8; ++r) {
      // D layout: VGPR r, lanes 0-15 -> M=r, N=lane; lanes 16-31 -> M=r+8
      const int m = r + sel * 8;
      float v0 = acc0[r] + bias;
      float v1 = acc1[r] + bias;
      if (RELU) { v0 = fmaxf(v0, 0.0f); v1 = fmaxf(v1, 0.0f); }
      if (TO_GLOBAL) {
        __builtin_nontemporal_store(
            v0, outp + (size_t)(t0 * 16 + m) * 128 + nt * 16 + ln);
        __builtin_nontemporal_store(
            v1, outp + (size_t)(t0 * 16 + 16 + m) * 128 + nt * 16 + ln);
      } else {
        s_act[swzc(t0 * 16 + m,      nt * 16 + ln)] = (_Float16)v0;
        s_act[swzc(t0 * 16 + 16 + m, nt * 16 + ln)] = (_Float16)v1;
      }
    }
  }
}

__global__ void __launch_bounds__(256, 1)
cobafa_field_kernel(Params p) {
  extern __shared__ __align__(16) char smem_raw[];
  _Float16* s_act = (_Float16*)smem_raw;          // 256 x 128 f16 = 64 KB
  _Float16* s_w   = s_act + 256 * 128;            // 128 x 128 f16 = 32 KB

  const int tid  = threadIdx.x;
  const int lane = tid & 31;
  const int wv   = tid >> 5;
  const int ln   = lane & 15;
  const int sel  = lane >> 4;

  // ---------------- Phase 1: per-point feature encoding -> LDS ----------------
  {
    const size_t pt = (size_t)blockIdx.x * 256 + tid;
    const float x = p.x[pt * 3 + 0];
    const float y = p.x[pt * 3 + 1];
    const float z = p.x[pt * 3 + 2];

    float coef[4];
    trilerp4<64>(p.coef, x, y, z, coef);

    float fe[16];
    {
      float v[4];
      trilerp4<32>(p.g0, enc1(x, 1.0f), enc1(y, 1.0f), enc1(z, 1.0f), v);
      fe[0] = v[0] * coef[0]; fe[1] = v[1] * coef[0];
      fe[2] = v[2] * coef[0]; fe[3] = v[3] * coef[0];
    }
    {
      float v[4];
      trilerp4<64>(p.g1, enc1(x, 2.0f), enc1(y, 2.0f), enc1(z, 2.0f), v);
      fe[4] = v[0] * coef[1]; fe[5] = v[1] * coef[1];
      fe[6] = v[2] * coef[1]; fe[7] = v[3] * coef[1];
    }
    {
      float v[4];
      trilerp4<128>(p.g2, enc1(x, 4.0f), enc1(y, 4.0f), enc1(z, 4.0f), v);
      fe[8]  = v[0] * coef[2]; fe[9]  = v[1] * coef[2];
      fe[10] = v[2] * coef[2]; fe[11] = v[3] * coef[2];
    }
    {
      float v[4];
      trilerp4<256>(p.g3, enc1(x, 8.0f), enc1(y, 8.0f), enc1(z, 8.0f), v);
      fe[12] = v[0] * coef[3]; fe[13] = v[1] * coef[3];
      fe[14] = v[2] * coef[3]; fe[15] = v[3] * coef[3];
    }

#pragma unroll
    for (int k = 0; k < 16; ++k) s_act[swzc(tid, k)] = (_Float16)fe[k];
#pragma unroll
    for (int k = 16; k < 32; ++k) s_act[swzc(tid, k)] = (_Float16)0.0f;  // K pad
  }

  // ---------------- Phase 2: 7-layer MLP on WMMA ----------------
  float* const outp = p.out + (size_t)blockIdx.x * 256 * 128;
  const int t0 = wv * 2;   // each wave owns 2 disjoint 16-point tiles

  for (int j = 0; j < 7; ++j) {
    // Cooperative transpose-load of W_j (global f32 [din,128]) into Wt[n][k] f16.
    const float* __restrict__ Wj = p.W[j];
    const int din = (j == 0) ? 16 : 128;
    for (int idx = tid; idx < din * 128; idx += 256) {
      int k = idx >> 7, n = idx & 127;
      s_w[swzc(n, k)] = (_Float16)Wj[idx];    // Wj[k*128+n] == Wj[idx]
    }
    if (j == 0) {
      for (int idx = tid; idx < 16 * 128; idx += 256) {   // zero K pad rows 16..31
        int k = 16 + (idx >> 7), n = idx & 127;
        s_w[swzc(n, k)] = (_Float16)0.0f;
      }
    }
    __syncthreads();  // weights ready; also orders previous layer's act writes

    const float* __restrict__ Bj = p.Bv[j];
    if (j == 0)      mlp_layer_pair<1, true,  false>(s_act, s_w, Bj, outp, t0, ln, sel);
    else if (j < 6)  mlp_layer_pair<4, true,  false>(s_act, s_w, Bj, outp, t0, ln, sel);
    else             mlp_layer_pair<4, false, true >(s_act, s_w, Bj, outp, t0, ln, sel);

    __syncthreads();  // protect s_w before next layer's overwrite
  }
}

extern "C" void kernel_launch(void* const* d_in, const int* in_sizes, int n_in,
                              void* d_out, int out_size, void* d_ws, size_t ws_size,
                              hipStream_t stream) {
  (void)n_in; (void)d_ws; (void)ws_size; (void)out_size;

  Params p;
  p.x    = (const float*)d_in[0];
  p.g0   = (const float*)d_in[1];
  p.g1   = (const float*)d_in[2];
  p.g2   = (const float*)d_in[3];
  p.g3   = (const float*)d_in[4];
  p.coef = (const float*)d_in[5];
  for (int j = 0; j < 7; ++j) {
    p.W[j]  = (const float*)d_in[6 + 2 * j];
    p.Bv[j] = (const float*)d_in[7 + 2 * j];
  }
  p.out = (float*)d_out;

  const int n = in_sizes[0] / 3;          // 524288
  const int blocks = n / 256;             // 2048 (N divisible by 256)
  constexpr unsigned SMEM_BYTES = (256 * 128 + 128 * 128) * 2;  // 96 KB

  cobafa_field_kernel<<<blocks, 256, SMEM_BYTES, stream>>>(p);
}
